// FFNO_34900904247501
// MI455X (gfx1250) — compile-verified, matrix operand order checked by reading, not run on hardware
//
#include <hip/hip_runtime.h>
#include <math.h>

// ---------------------------------------------------------------------------
// FFNO forward for MI455X (gfx1250, wave32, WMMA).
// Memory-bound pipeline -> activations resident in BF16 (halves HBM traffic;
// WMMA inputs were bf16-rounded anyway), fp32 WMMA accumulation.
// All heavy math as WMMA GEMMs with ALIGNED shapes (M%64, N%16, K%32 == 0).
// Strides are TEMPLATE CONSTANTS, offsets unsigned 32-bit -> saddr+voffset
// addressing. One wave computes a 64x16 C strip: 4 M-tiles share one B frag.
// Encoder (K=3) and decoder (M=1) are memory-bound VALU kernels.
// ---------------------------------------------------------------------------

typedef __bf16 bf;
typedef __attribute__((ext_vector_type(16))) __bf16 v16bf;
typedef __attribute__((ext_vector_type(8)))  __bf16 v8bf;
typedef __attribute__((ext_vector_type(8)))  float  v8f;

#define F_ACCUM 1
#define F_BIAS  2
#define F_GELU  4

// Branch-free GELU (tanh approx, matches jax.nn.gelu approximate=True).
__device__ __forceinline__ float gelu_tanh(float x) {
  float u = 0.7978845608028654f * (x + 0.044715f * x * x * x);
  float e = __builtin_amdgcn_exp2f(2.8853900817779268f * u);  // exp(2u)
  float t = 1.0f - 2.0f * __builtin_amdgcn_rcpf(e + 1.0f);    // tanh(u)
  return 0.5f * x * (1.0f + t);
}

// ISA 16-bit A/B fragment K-offset for element j, lane-half h:
//   j<8 -> j + 8h ; j>=8 -> j + 8 + 8h
__device__ __forceinline__ int kof(int j, int half) {
  return (j < 8) ? (j + 8 * half) : (j + 8 + 8 * half);
}

// Gather one bf16 fragment; element j at base[fixed + (kb+kof(j))*KS].
// KS==1 fast path: j=0..7 and j=8..15 are contiguous K runs -> 2x b128 loads
// (all call sites keep `fixed` a multiple of 8 elements => 16B aligned).
template <int KS>
__device__ __forceinline__ v16bf load_frag(const bf* __restrict__ base,
                                           unsigned fixed, int kb, int half) {
  if constexpr (KS == 1) {
    unsigned o = fixed + (unsigned)kb + 8u * (unsigned)half;
    v8bf q0 = *(const v8bf*)(base + o);        // j = 0..7  (k = kb+8h ..)
    v8bf q1 = *(const v8bf*)(base + o + 16u);  // j = 8..15 (k = kb+16+8h ..)
    return __builtin_shufflevector(q0, q1, 0, 1, 2, 3, 4, 5, 6, 7,
                                   8, 9, 10, 11, 12, 13, 14, 15);
  } else {
    v16bf f;
#pragma unroll
    for (int j = 0; j < 16; ++j)
      f[j] = base[fixed + (unsigned)(kb + kof(j, half)) * (unsigned)KS];
    return f;
  }
}

// One wave -> 64x16 tile strip of C = alpha*(A@B) [+bias][gelu][+=C].
// Requires M%64==0, N%16==0, K%32==0. All strides compile-time.
// Row addressing for A and C uses a split stride (hi for r>>9, lo for r&511)
// so [P,H,W] tensors flattened over (p,spatial) rows stay addressable.
template <int FLAGS, int ARSHI, int ARSLO, int ACS, int BRS, int BCS,
          int CRSHI, int CRSLO, int CCS, int BSA, int BSB, int BSC>
__global__ __launch_bounds__(32) void wmma_gemm(
    const bf* __restrict__ A, const bf* __restrict__ B,
    bf* __restrict__ C, int K, const float* __restrict__ bias, float alpha)
{
  const unsigned z = blockIdx.z;
  const unsigned zA = z * (unsigned)BSA;
  const unsigned zB = z * (unsigned)BSB;
  const unsigned zC = z * (unsigned)BSC;

  const int lane = threadIdx.x;      // 0..31 (wave32)
  const int l16  = lane & 15;
  const int half = lane >> 4;
  const int m0 = blockIdx.x * 64;    // 4 vertical tiles per wave
  const int n0 = blockIdx.y * 16;

  v8f acc0 = {0.f,0.f,0.f,0.f,0.f,0.f,0.f,0.f};
  v8f acc1 = acc0, acc2 = acc0, acc3 = acc0;

  unsigned arow[4];
#pragma unroll
  for (int mt = 0; mt < 4; ++mt) {
    unsigned am = (unsigned)(m0 + mt * 16 + l16);
    arow[mt] = zA + (am >> 9) * (unsigned)ARSHI + (am & 511u) * (unsigned)ARSLO;
  }
  const unsigned bcol = zB + (unsigned)(n0 + l16) * (unsigned)BCS;

#pragma unroll 1
  for (int kb = 0; kb < K; kb += 32) {
    // Fragment-at-a-time: low register liveness, loads still overlap WMMAs.
    v16bf bfrag = load_frag<BRS>(B, bcol, kb, half);
    v16bf a0 = load_frag<ACS>(A, arow[0], kb, half);
    acc0 = __builtin_amdgcn_wmma_f32_16x16x32_bf16(false, a0, false, bfrag, (short)0, acc0, false, false);
    v16bf a1 = load_frag<ACS>(A, arow[1], kb, half);
    acc1 = __builtin_amdgcn_wmma_f32_16x16x32_bf16(false, a1, false, bfrag, (short)0, acc1, false, false);
    v16bf a2 = load_frag<ACS>(A, arow[2], kb, half);
    acc2 = __builtin_amdgcn_wmma_f32_16x16x32_bf16(false, a2, false, bfrag, (short)0, acc2, false, false);
    v16bf a3 = load_frag<ACS>(A, arow[3], kb, half);
    acc3 = __builtin_amdgcn_wmma_f32_16x16x32_bf16(false, a3, false, bfrag, (short)0, acc3, false, false);
  }

  // Straight-line epilogue (shapes aligned, FLAGS compile-time).
  const unsigned cn = zC + (unsigned)(n0 + l16) * (unsigned)CCS;
#pragma unroll
  for (int mt = 0; mt < 4; ++mt) {
    v8f acc = (mt == 0) ? acc0 : (mt == 1) ? acc1 : (mt == 2) ? acc2 : acc3;
#pragma unroll
    for (int e = 0; e < 8; ++e) {
      unsigned m = (unsigned)(m0 + mt * 16 + e + 8 * half);  // C/D row for elem e
      float r = alpha * acc[e];
      if (FLAGS & F_BIAS) r += bias[m];
      if (FLAGS & F_GELU) r = gelu_tanh(r);
      unsigned caddr = (m >> 9) * (unsigned)CRSHI + (m & 511u) * (unsigned)CRSLO + cn;
      if (FLAGS & F_ACCUM) r += (float)C[caddr];
      C[caddr] = (bf)r;
    }
  }
}

// Build truncated DFT matrices (bf16).
// Bf[x*32+kk]: kk<16 -> cos(2*pi*k*x/512), kk>=16 -> -sin  (forward rfft, 16 modes)
// Bi[kk*512+x]: c_k*cos / -c_k*sin with c_0=1/512, c_k=2/512  (truncated irfft)
__global__ void build_dft(bf* __restrict__ Bf, bf* __restrict__ Bi) {
  int t = blockIdx.x * blockDim.x + threadIdx.x;
  if (t >= 512 * 32) return;
  int xx = t >> 5, kk = t & 31, k = kk & 15;
  int ph = (k * xx) & 511;
  float th = 6.283185307179586f * (float)ph * (1.0f / 512.0f);
  float c = cosf(th), sn = sinf(th);
  Bf[xx * 32 + kk] = (bf)((kk < 16) ? c : -sn);
  float scale = (k == 0) ? (1.0f / 512.0f) : (2.0f / 512.0f);
  Bi[kk * 512 + xx] = (bf)((kk < 16) ? scale * c : -scale * sn);
}

// fp32 -> bf16 weight conversion (one-shot, tiny).
__global__ void f2bf_kernel(const float* __restrict__ src, bf* __restrict__ dst, int n) {
  int i = blockIdx.x * blockDim.x + threadIdx.x;
  if (i < n) dst[i] = (bf)src[i];
}

// Encoder: h[o,n] = Wenc[o,0]*x0[n] + Wenc[o,1]*x1[n] + Wenc[o,2]*u[n] + b[o]
__global__ void encode_kernel(const float* __restrict__ x, const float* __restrict__ u,
                              const float* __restrict__ W, const float* __restrict__ b,
                              bf* __restrict__ h) {
  int i = blockIdx.x * blockDim.x + threadIdx.x;  // 64 * 262144 threads
  int o = i >> 18;
  int n = i & 262143;
  h[i] = (bf)(W[o * 3 + 0] * x[n] + W[o * 3 + 1] * x[262144 + n]
            + W[o * 3 + 2] * u[n] + b[o]);
}

// Decoder: out[n] = sum_i Wdec[i]*h[i,n] + bdec   (M=1; memory-bound VALU)
__global__ void decode_kernel(const bf* __restrict__ h, const float* __restrict__ W,
                              const float* __restrict__ b, float* __restrict__ out) {
  int n = blockIdx.x * blockDim.x + threadIdx.x;  // 262144 threads
  float acc = b[0];
#pragma unroll
  for (int i = 0; i < 64; ++i) acc += W[i] * (float)h[(unsigned)i * 262144u + n];
  out[n] = acc;
}

// ---------------------------------------------------------------------------

extern "C" void kernel_launch(void* const* d_in, const int* in_sizes, int n_in,
                              void* d_out, int out_size, void* d_ws, size_t ws_size,
                              hipStream_t stream) {
  (void)in_sizes; (void)n_in; (void)out_size; (void)ws_size;

  const float* u    = (const float*)d_in[0];   // [1,512,512]
  const float* x    = (const float*)d_in[1];   // [2,512,512]
  const float* Wenc = (const float*)d_in[2];   // [64,3]
  const float* benc = (const float*)d_in[3];   // [64]
  const float* W1   = (const float*)d_in[4];   // [4,64,64]
  const float* b1   = (const float*)d_in[5];   // [4,64]
  const float* W2   = (const float*)d_in[6];   // [4,64,64]
  const float* b2   = (const float*)d_in[7];   // [4,64]
  const float* Are  = (const float*)d_in[8];   // [4,64,64,16,2]
  const float* Aim  = (const float*)d_in[9];   // [4,64,64,16,2]
  const float* Wdec = (const float*)d_in[10];  // [1,64]
  const float* bdec = (const float*)d_in[11];  // [1]
  float* out = (float*)d_out;                  // [1,512,512]

  const long long PHW  = 16777216;   // 64*512*512
  const long long SLAB = 32768;      // 64*512  (one mode slab / one row-group)

  bf* ws   = (bf*)d_ws;
  bf* h    = ws;                     // [64,512,512] bf16
  bf* s    = h   + PHW;              // spectral output
  bf* t    = s   + PHW;              // MLP hidden
  bf* V    = t   + PHW;              // DFT coeffs  [32 slabs][64][512]
  bf* R    = V   + 32 * SLAB;        // mixed coeffs [32 slabs][64][512]
  bf* Bf   = R   + 32 * SLAB;        // [512,32] forward DFT matrix
  bf* Bi   = Bf  + 512 * 32;         // [32,512] inverse DFT matrix
  bf* W1b  = Bi  + 512 * 32;         // [4,64,64] bf16 weights
  bf* W2b  = W1b + 16384;
  bf* Areb = W2b + 16384;            // [4,64,64,16,2] bf16
  bf* Aimb = Areb + 524288;

  build_dft<<<(512 * 32 + 255) / 256, 256, 0, stream>>>(Bf, Bi);
  f2bf_kernel<<<64, 256, 0, stream>>>(W1, W1b, 16384);
  f2bf_kernel<<<64, 256, 0, stream>>>(W2, W2b, 16384);
  f2bf_kernel<<<2048, 256, 0, stream>>>(Are, Areb, 524288);
  f2bf_kernel<<<2048, 256, 0, stream>>>(Aim, Aimb, 524288);
  encode_kernel<<<(unsigned)(PHW / 256), 256, 0, stream>>>(x, u, Wenc, benc, h);

  for (int l = 0; l < 4; ++l) {
    for (int a = 0; a < 2; ++a) {
      // ---- forward truncated DFT: V[kk][p][col] = sum_axis h * Bf ----
      if (a == 0) {
        // along y; rows r=(p,x): addr = p*262144 + x (+ y*512)
        wmma_gemm<0, 262144, 1, 512, 32, 1, 512, 1, 32768, 0, 0, 0>
          <<<dim3(512, 2, 1), 32, 0, stream>>>(h, Bf, V, 512, nullptr, 1.0f);
      } else {
        // along x; rows r=(p,y): addr = p*262144 + y*512 (+ x)
        wmma_gemm<0, 262144, 512, 1, 32, 1, 512, 1, 32768, 0, 0, 0>
          <<<dim3(512, 2, 1), 32, 0, stream>>>(h, Bf, V, 512, nullptr, 1.0f);
      }

      // ---- per-mode complex channel mix: R = A_cplx @ V_cplx (z = mode) ----
      const bf* Ar = Areb + l * 131072 + a;  // A[o,i,m,axis]: o*2048+i*32+m*2+a
      const bf* Ai = Aimb + l * 131072 + a;
      // Re(R) = Ar@Vr - Ai@Vi
      wmma_gemm<0,       0, 2048, 32, 512, 1, 0, 512, 1, 2, 32768, 32768>
        <<<dim3(1, 32, 16), 32, 0, stream>>>(Ar, V, R, 64, nullptr, 1.0f);
      wmma_gemm<F_ACCUM, 0, 2048, 32, 512, 1, 0, 512, 1, 2, 32768, 32768>
        <<<dim3(1, 32, 16), 32, 0, stream>>>(Ai, V + 16 * SLAB, R, 64, nullptr, -1.0f);
      // Im(R) = Ar@Vi + Ai@Vr
      wmma_gemm<0,       0, 2048, 32, 512, 1, 0, 512, 1, 2, 32768, 32768>
        <<<dim3(1, 32, 16), 32, 0, stream>>>(Ar, V + 16 * SLAB, R + 16 * SLAB, 64, nullptr, 1.0f);
      wmma_gemm<F_ACCUM, 0, 2048, 32, 512, 1, 0, 512, 1, 2, 32768, 32768>
        <<<dim3(1, 32, 16), 32, 0, stream>>>(Ai, V, R + 16 * SLAB, 64, nullptr, 1.0f);

      // ---- inverse truncated DFT: s (+)= R @ Bi ----
      if (a == 0) {
        wmma_gemm<0,       512, 1, 32768, 512, 1, 262144, 1, 512, 0, 0, 0>
          <<<dim3(512, 32, 1), 32, 0, stream>>>(R, Bi, s, 32, nullptr, 1.0f);   // store (init s)
      } else {
        wmma_gemm<F_ACCUM, 512, 1, 32768, 512, 1, 262144, 512, 1, 0, 0, 0>
          <<<dim3(512, 32, 1), 32, 0, stream>>>(R, Bi, s, 32, nullptr, 1.0f);   // += axis-x part
      }
    }

    // ---- t = gelu(W1 @ s + b1) ----
    wmma_gemm<F_BIAS | F_GELU, 0, 64, 1, 262144, 1, 0, 262144, 1, 0, 0, 0>
      <<<dim3(1, 16384, 1), 32, 0, stream>>>(W1b + l * 4096, s, t,
                                             64, b1 + l * 64, 1.0f);
    // ---- h += gelu(W2 @ t + b2)  (residual) ----
    wmma_gemm<F_BIAS | F_GELU | F_ACCUM, 0, 64, 1, 262144, 1, 0, 262144, 1, 0, 0, 0>
      <<<dim3(1, 16384, 1), 32, 0, stream>>>(W2b + l * 4096, t, h,
                                             64, b2 + l * 64, 1.0f);
  }

  // ---- decoder: out = Wdec @ h + bdec ----
  decode_kernel<<<(unsigned)(262144 / 256), 256, 0, stream>>>(h, Wdec, bdec, out);
}
